// SuperedgeLearn_68143951118804
// MI455X (gfx1250) — matrix operand order repaired and checked by reading, not compiled, not run on hardware
//
#include <hip/hip_runtime.h>
#include <hip/hip_bf16.h>
#include <stdint.h>

// ---------------------------------------------------------------------------
// SuperedgeLearn, MI455X (gfx1250).  B=8192 pairs, 2N=64 neighbors, F=E=H=128.
// Algebraic reassociation makes this memory-bound (~554MB @ 23.3TB/s):
//   scores = X @ (Wk @ q),  local = (alpha^T X) @ Wv
// Dense projections run as bf16 WMMA (16 pairs batched per workgroup);
// neighbor features stream through LDS via async global->LDS copies.
// ---------------------------------------------------------------------------

typedef __attribute__((ext_vector_type(16))) __bf16 v16bf;
typedef __attribute__((ext_vector_type(8)))  float  v8f;

#define XB_PITCH 260                      // floats; 1040B = 65*16B (16B aligned rows, bank skew)
#define XB_BUF   (64 * XB_PITCH)          // floats per buffer
#define OFF_XB   0
#define OFF_X0   (2 * XB_BUF * 4)         // 133120
#define OFF_EE   (OFF_X0 + 16 * 256 * 2)  // 141312
#define OFF_Q    (OFF_EE + 16 * 128 * 2)  // 145408
#define OFF_T    (OFF_Q  + 16 * 128 * 2)  // 149504
#define OFF_Y    (OFF_T  + 16 * 256 * 4)  // 165888
#define OFF_SC   (OFF_Y  + 16 * 256 * 2)  // 174080
#define OFF_AL   (OFF_SC + 64 * 4 * 4)    // 175104
#define SMEM_BYTES (OFF_AL + 64 * 4)      // 175360  (< 320KB)

// ws layout (bf16 elements)
#define WS_WEDGE 0        // WeT [128 n][256 k]   (W_edge transposed, n-major)
#define WS_WQ    32768    // WqT [128 n][128 k]
#define WS_WK    49152    // WkB [256 n=e][128 k=h] == Wk row-major in bf16
#define WS_WV    81920    // WvT [128 n][256 k]

union FragU { uint4 u[2]; v16bf v; };

// A-fragment (16x32 bf16) from LDS, ISA layout:
// lane<16: K = {0..7, 16..23}; lane>=16: K = {8..15, 24..31}; M = lane&15
__device__ __forceinline__ v16bf afrag_lds(const __bf16* tile, int stride, int lane) {
  int m = lane & 15, hh = lane >> 4;
  const __bf16* r = tile + m * stride + hh * 8;
  FragU f;
  f.u[0] = *(const uint4*)(r);
  f.u[1] = *(const uint4*)(r + 16);
  return f.v;
}

// B-fragment (32x16 bf16) from global, weights stored n-major (w[n][k]):
// lane = column N = lane&15; lane<16: K = {0..7, 16..23}; lane>=16: K = {8..15, 24..31}.
__device__ __forceinline__ v16bf bfrag_glb(const __bf16* w, int stride, int n0, int k0, int lane) {
  int n = lane & 15, hh = lane >> 4;
  const __bf16* p = w + (size_t)(n0 + n) * stride + k0 + hh * 8;
  FragU f;
  f.u[0] = *(const uint4*)(p);
  f.u[1] = *(const uint4*)(p + 16);
  return f.v;
}

__device__ __forceinline__ uint32_t lds_byte_off(const void* p) {
  return (uint32_t)(uintptr_t)(__attribute__((address_space(3))) const void*)p;
}

// Issue 16 async b128 copies per wave: one 64x256-f32 neighbor tile -> LDS.
__device__ __forceinline__ void issue_xtile(int bglob, uint32_t lds_base_off,
                                            const float* mrel0, const float* drel0,
                                            const float* mnode1, const float* dnode1, int tid) {
  size_t base = (size_t)bglob * 32 * 128;
#pragma unroll
  for (int i = 0; i < 16; ++i) {
    int c  = tid + i * 256;          // 0..4095 sixteen-byte chunks
    int n  = c >> 6;                 // neighbor row 0..63
    int e4 = c & 63;                 // 16B chunk within row
    int nn = (n < 32) ? n : (n - 32);
    int ee = (e4 < 32) ? (e4 * 4) : ((e4 - 32) * 4);
    const float* rel = (n < 32) ? mrel0 : drel0;
    const float* nod = (n < 32) ? mnode1 : dnode1;
    const float* src = ((e4 < 32) ? rel : nod) + base + (size_t)nn * 128 + ee;
    uint32_t loff = lds_base_off + (uint32_t)((n * XB_PITCH + e4 * 4) * 4);
    asm volatile("global_load_async_to_lds_b128 %0, %1, off"
                 :: "v"(loff), "v"((unsigned long long)(uintptr_t)src)
                 : "memory");
  }
}

__global__ void superedge_prep_weights(const float* __restrict__ We, const float* __restrict__ Wq,
                                       const float* __restrict__ Wk, const float* __restrict__ Wv,
                                       __bf16* __restrict__ ws) {
  int t = blockIdx.x * blockDim.x + threadIdx.x;
  int stride = gridDim.x * blockDim.x;
  __bf16* WeT = ws + WS_WEDGE;
  __bf16* WqT = ws + WS_WQ;
  __bf16* WkB = ws + WS_WK;
  __bf16* WvT = ws + WS_WV;
  for (int i = t; i < 128 * 256; i += stride) { int n = i >> 8, k = i & 255; WeT[i] = (__bf16)We[k * 128 + n]; }
  for (int i = t; i < 128 * 128; i += stride) { int n = i >> 7, k = i & 127; WqT[i] = (__bf16)Wq[k * 128 + n]; }
  for (int i = t; i < 256 * 128; i += stride) { WkB[i] = (__bf16)Wk[i]; }
  for (int i = t; i < 128 * 256; i += stride) { int n = i >> 8, k = i & 255; WvT[i] = (__bf16)Wv[k * 128 + n]; }
}

__global__ void superedge_main(const float* __restrict__ mnode0, const float* __restrict__ dnode0,
                               const float* __restrict__ mnode1, const float* __restrict__ dnode1,
                               const float* __restrict__ mrel0,  const float* __restrict__ drel0,
                               const float* __restrict__ b_edge, const __bf16* __restrict__ ws,
                               float* __restrict__ out) {
  extern __shared__ char smem[];
  float*  Xb  = (float*)(smem + OFF_XB);
  __bf16* X0  = (__bf16*)(smem + OFF_X0);
  __bf16* EE  = (__bf16*)(smem + OFF_EE);
  __bf16* Qs  = (__bf16*)(smem + OFF_Q);
  float*  Ts  = (float*)(smem + OFF_T);
  __bf16* Y16 = (__bf16*)(smem + OFF_Y);
  float*  sc4 = (float*)(smem + OFF_SC);
  float*  alp = (float*)(smem + OFF_AL);

  const __bf16* WeT = ws + WS_WEDGE;
  const __bf16* WqT = ws + WS_WQ;
  const __bf16* WkB = ws + WS_WK;
  const __bf16* WvT = ws + WS_WV;

  const int tid  = threadIdx.x;
  const int lane = tid & 31;
  const int wv   = tid >> 5;
  const int b0   = blockIdx.x * 16;
  const uint32_t xb_off = lds_byte_off(Xb);

  // Prefetch neighbor tiles for local pairs 0 and 1 (hidden under WMMA stages 1-3)
  issue_xtile(b0 + 0, xb_off,               mrel0, drel0, mnode1, dnode1, tid);
  issue_xtile(b0 + 1, xb_off + XB_BUF * 4,  mrel0, drel0, mnode1, dnode1, tid);

  // Stage 0: center features [m0 || d0] -> X0 bf16 [16][256]
#pragma unroll
  for (int i = 0; i < 16; ++i) {
    int idx = tid + i * 256;
    int r = idx >> 8, e = idx & 255;
    float v = (e < 128) ? mnode0[(size_t)(b0 + r) * 128 + e]
                        : dnode0[(size_t)(b0 + r) * 128 + (e - 128)];
    X0[r * 256 + e] = (__bf16)v;
  }
  __syncthreads();

  // Stage 1 (WMMA): EE = relu(X0 @ We + b)  [16x256]@[256x128]
  {
    v8f acc = {};
    int n0 = wv * 16;
#pragma unroll
    for (int kt = 0; kt < 8; ++kt) {
      v16bf a  = afrag_lds(X0 + kt * 32, 256, lane);
      v16bf bm = bfrag_glb(WeT, 256, n0, kt * 32, lane);
      acc = __builtin_amdgcn_wmma_f32_16x16x32_bf16(false, a, false, bm, (short)0, acc, false, false);
    }
    int nn = n0 + (lane & 15), hh = lane >> 4;
    float bias = b_edge[nn];
#pragma unroll
    for (int r = 0; r < 8; ++r) {
      int M = r + hh * 8;
      float val = fmaxf(acc[r] + bias, 0.0f);
      out[(size_t)(b0 + M) * 256 + nn] = val;      // output cols 0..127
      EE[M * 128 + nn] = (__bf16)val;
    }
  }
  __syncthreads();

  // Stage 2 (WMMA): Q = EE @ Wq  [16x128]@[128x128]
  {
    v8f acc = {};
    int n0 = wv * 16;
#pragma unroll
    for (int kt = 0; kt < 4; ++kt) {
      v16bf a  = afrag_lds(EE + kt * 32, 128, lane);
      v16bf bm = bfrag_glb(WqT, 128, n0, kt * 32, lane);
      acc = __builtin_amdgcn_wmma_f32_16x16x32_bf16(false, a, false, bm, (short)0, acc, false, false);
    }
    int nn = n0 + (lane & 15), hh = lane >> 4;
#pragma unroll
    for (int r = 0; r < 8; ++r) Qs[(r + hh * 8) * 128 + nn] = (__bf16)acc[r];
  }
  __syncthreads();

  // Stage 3 (WMMA): T = Q @ Wk^T  [16x128]@[128x256] -> f32
#pragma unroll
  for (int ht = 0; ht < 2; ++ht) {
    int nt = wv + ht * 8;
    v8f acc = {};
    int n0 = nt * 16;
#pragma unroll
    for (int kt = 0; kt < 4; ++kt) {
      v16bf a  = afrag_lds(Qs + kt * 32, 128, lane);
      v16bf bm = bfrag_glb(WkB, 128, n0, kt * 32, lane);
      acc = __builtin_amdgcn_wmma_f32_16x16x32_bf16(false, a, false, bm, (short)0, acc, false, false);
    }
    int nn = n0 + (lane & 15), hh = lane >> 4;
#pragma unroll
    for (int r = 0; r < 8; ++r) Ts[(r + hh * 8) * 256 + nn] = acc[r];
  }
  __syncthreads();

  // Stage 4: per-pair attention over LDS-staged X (double-buffered async)
  for (int b = 0; b < 16; ++b) {
    if (b < 15) asm volatile("s_wait_asynccnt 16" ::: "memory");  // oldest tile done
    else        asm volatile("s_wait_asynccnt 0"  ::: "memory");
    __syncthreads();
    const float* X = Xb + (b & 1) * XB_BUF;

    // scores partials: 64 rows x 4 chunks of 64
    {
      int n = tid & 63, c = tid >> 6;
      const float* xr = X + n * XB_PITCH + c * 64;
      const float* tr = Ts + b * 256 + c * 64;
      float p = 0.0f;
#pragma unroll
      for (int e = 0; e < 64; ++e) p = __builtin_fmaf(xr[e], tr[e], p);
      sc4[n * 4 + c] = p;
    }
    __syncthreads();

    // softmax over 64 scores on wave 0 (shuffle reductions)
    if (tid < 32) {
      const float scale = 0.08838834764831845f;  // 1/sqrt(128)
      int l2 = lane + 32;
      float sa = (sc4[lane * 4] + sc4[lane * 4 + 1] + sc4[lane * 4 + 2] + sc4[lane * 4 + 3]) * scale;
      float sb = (sc4[l2 * 4]   + sc4[l2 * 4 + 1]   + sc4[l2 * 4 + 2]   + sc4[l2 * 4 + 3])   * scale;
      float m = fmaxf(sa, sb);
#pragma unroll
      for (int off = 16; off > 0; off >>= 1) m = fmaxf(m, __shfl_xor(m, off, 32));
      float ea = __expf(sa - m), eb = __expf(sb - m);
      float s = ea + eb;
#pragma unroll
      for (int off = 16; off > 0; off >>= 1) s += __shfl_xor(s, off, 32);
      float inv = 1.0f / s;
      alp[lane] = ea * inv;
      alp[l2]   = eb * inv;
    }
    __syncthreads();

    // y = alpha^T X  -> Y16 row (bf16)
    {
      float acc = 0.0f;
#pragma unroll
      for (int n = 0; n < 64; ++n) acc = __builtin_fmaf(alp[n], X[n * XB_PITCH + tid], acc);
      Y16[b * 256 + tid] = (__bf16)acc;
    }
    __syncthreads();

    if (b + 2 < 16)
      issue_xtile(b0 + b + 2, xb_off + (uint32_t)((b & 1) * XB_BUF * 4),
                  mrel0, drel0, mnode1, dnode1, tid);
  }

  // Stage 5 (WMMA): OUT2 = Y @ Wv  [16x256]@[256x128] -> output cols 128..255
  {
    v8f acc = {};
    int n0 = wv * 16;
#pragma unroll
    for (int kt = 0; kt < 8; ++kt) {
      v16bf a  = afrag_lds(Y16 + kt * 32, 256, lane);
      v16bf bm = bfrag_glb(WvT, 256, n0, kt * 32, lane);
      acc = __builtin_amdgcn_wmma_f32_16x16x32_bf16(false, a, false, bm, (short)0, acc, false, false);
    }
    int nn = n0 + (lane & 15), hh = lane >> 4;
#pragma unroll
    for (int r = 0; r < 8; ++r)
      out[(size_t)(b0 + r + hh * 8) * 256 + 128 + nn] = acc[r];
  }
}

extern "C" void kernel_launch(void* const* d_in, const int* in_sizes, int n_in,
                              void* d_out, int out_size, void* d_ws, size_t ws_size,
                              hipStream_t stream) {
  const float* mnode0 = (const float*)d_in[0];
  const float* dnode0 = (const float*)d_in[1];
  const float* mnode1 = (const float*)d_in[2];
  const float* dnode1 = (const float*)d_in[3];
  const float* mrel0  = (const float*)d_in[4];
  const float* drel0  = (const float*)d_in[5];
  const float* W_edge = (const float*)d_in[6];
  const float* b_edge = (const float*)d_in[7];
  const float* Wq     = (const float*)d_in[8];
  const float* Wk     = (const float*)d_in[9];
  const float* Wv     = (const float*)d_in[10];
  __bf16* ws = (__bf16*)d_ws;
  float* out = (float*)d_out;

  hipLaunchKernelGGL(superedge_prep_weights, dim3(120), dim3(256), 0, stream,
                     W_edge, Wq, Wk, Wv, ws);
  hipLaunchKernelGGL(superedge_main, dim3(8192 / 16), dim3(256), SMEM_BYTES, stream,
                     mnode0, dnode0, mnode1, dnode1, mrel0, drel0, b_edge, ws, out);
}
